// LanguageModelAttention_75892072121044
// MI455X (gfx1250) — matrix operand order repaired and checked by reading, not compiled, not run on hardware
//
#include <hip/hip_runtime.h>
#include <hip/hip_bf16.h>
#include <stdint.h>

// Model dims
#define VV 32000
#define DD 512
#define LL 4
#define NN 1024
#define HH 8
#define BB 2
#define SS 128
#define DHH 64
#define PHI_F 1.6180339887498949f
#define INV_SCALE 0.08838834764831845f   // 1/sqrt(2*DH) = 1/sqrt(128)

typedef __attribute__((ext_vector_type(16))) _Float16 v16h;
typedef __attribute__((ext_vector_type(8)))  _Float16 v8h;
typedef __attribute__((ext_vector_type(8)))  float    v8f;

// ---------------------------------------------------------------------------
// One-time: out_W fp32 -> fp16 (halves per-step traffic; 32.75MB fits in L2)
// ---------------------------------------------------------------------------
__global__ void convert_w_kernel(const float* __restrict__ W, _Float16* __restrict__ Wh, int n) {
    int idx = blockIdx.x * 256 + threadIdx.x;
    if (idx < n) Wh[idx] = (_Float16)W[idx];
}

__global__ void init_zero_kernel(float* __restrict__ p, int n) {
    int idx = blockIdx.x * 256 + threadIdx.x;
    if (idx < n) p[idx] = 0.0f;
}

// ---------------------------------------------------------------------------
// Oscillator step: h_real/h_imag rotation, x = h_real + h_imag
// ---------------------------------------------------------------------------
__global__ void osc_kernel(const int* __restrict__ ids, const float* __restrict__ temb,
                           float* __restrict__ hre, float* __restrict__ him,
                           float* __restrict__ x, int t, const int* __restrict__ tstart) {
    int idx = blockIdx.x * 256 + threadIdx.x;             // [0, B*D)
    if (idx >= BB * DD) return;
    float t_phi = (float)(t + *tstart) * PHI_F;
    int b = idx >> 9, d = idx & (DD - 1);
    int tok = ids[b * SS + t];
    float w_t = temb[(size_t)tok * (2 * DD) + d];
    float b_t = temb[(size_t)tok * (2 * DD) + DD + d];
    float wl = 1.0f + fabsf(w_t);
    float th_r = hre[idx] / wl + b_t + t_phi;
    float th_i = him[idx] / wl + b_t + t_phi;
    float sr = __sinf(th_r), cr = __cosf(th_r);
    float si = __sinf(th_i), ci = __cosf(th_i);
    float nr = cr * ci - sr * si;
    float ni = cr * si + sr * ci;
    hre[idx] = nr;
    him[idx] = ni;
    x[idx] = nr + ni;
}

// ---------------------------------------------------------------------------
// Attention + projection + LayerNorm for one layer. One block per batch (B=2),
// 256 threads. Wave32-level softmax via __shfl_xor (one wave per head).
// ---------------------------------------------------------------------------
__global__ void attn_kernel(const float* __restrict__ wq, const float* __restrict__ bq,
                            const float* __restrict__ wk, const float* __restrict__ bk,
                            const float* __restrict__ attn_proj,
                            const float* __restrict__ ln_g, const float* __restrict__ ln_b,
                            float* __restrict__ x, float* __restrict__ cache,
                            int l, int t, const int* __restrict__ tstart) {
    __shared__ float sq[HH * 2 * DHH];   // per-head [cos|sin] q features
    __shared__ float ssc[HH * SS];       // scores -> weights
    __shared__ float swsum[SS];          // sum over heads of weights
    __shared__ float sctx[DD];
    __shared__ float sx[DD];
    __shared__ float red[256];

    const int b = blockIdx.x, tid = threadIdx.x;
    const float t_phi = (float)(t + *tstart) * PHI_F;
    float* cb = cache + ((size_t)l * BB + b) * SS * DD;   // cache[l,b,s,d]

    // 1. write this step's state into the cache; stage x in LDS
    for (int d = tid; d < DD; d += 256) {
        float v = x[b * DD + d];
        cb[(size_t)t * DD + d] = v;
        sx[d] = v;
    }
    __syncthreads();

    // 2. q features
    for (int e = tid; e < DD; e += 256) {
        int h = e >> 6, i = e & (DHH - 1);
        float tq = sx[e] / (1.0f + fabsf(wq[(l * HH + h) * DHH + i]))
                 + bq[(l * HH + h) * DHH + i] + t_phi;
        sq[h * 128 + i]      = __cosf(tq);
        sq[h * 128 + 64 + i] = __sinf(tq);
    }
    __syncthreads();

    float xp0, xp1;   // projected x for d = tid, tid+256
    if (t > 0) {
        // 3. scores over (h, s<t)
        for (int p = tid; p < HH * t; p += 256) {
            int h = p / t, s = p - h * t;
            const float* st  = cb + (size_t)s * DD + h * DHH;
            const float* wkl = wk + (l * HH + h) * DHH;
            const float* bkl = bk + (l * HH + h) * DHH;
            const float* qh  = sq + h * 128;
            float acc = 0.0f;
            for (int i = 0; i < DHH; ++i) {
                float tk = st[i] / (1.0f + fabsf(wkl[i])) + bkl[i];
                acc += qh[i] * __cosf(tk) + qh[64 + i] * __sinf(tk);
            }
            ssc[h * SS + s] = acc * INV_SCALE;
        }
        __syncthreads();

        // 4. softmax per head, one wave32 per head
        {
            int wave = tid >> 5, lane = tid & 31;
            if (wave < HH) {
                float m = -1e30f;
                for (int s = lane; s < t; s += 32) m = fmaxf(m, ssc[wave * SS + s]);
                for (int off = 16; off > 0; off >>= 1) m = fmaxf(m, __shfl_xor(m, off, 32));
                float sum = 0.0f;
                for (int s = lane; s < t; s += 32) {
                    float e = __expf(ssc[wave * SS + s] - m);
                    ssc[wave * SS + s] = e;
                    sum += e;
                }
                for (int off = 16; off > 0; off >>= 1) sum += __shfl_xor(sum, off, 32);
                float inv = 1.0f / sum;
                for (int s = lane; s < t; s += 32) ssc[wave * SS + s] *= inv;
            }
        }
        __syncthreads();

        // 5. collapse heads: wsum[s] = sum_h w[h,s]
        for (int s = tid; s < t; s += 256) {
            float acc = 0.0f;
            for (int h = 0; h < HH; ++h) acc += ssc[h * SS + s];
            swsum[s] = acc;
        }
        __syncthreads();

        // 6. ctx[d] = sum_s wsum[s] * states[s,d]
        for (int d = tid; d < DD; d += 256) {
            float acc = 0.0f;
            for (int s = 0; s < t; ++s) acc += swsum[s] * cb[(size_t)s * DD + d];
            sctx[d] = acc;
        }
        __syncthreads();

        // 7. x += ctx @ attn_proj^T
        {
            const float* ap0 = attn_proj + ((size_t)l * DD + tid) * DD;
            const float* ap1 = attn_proj + ((size_t)l * DD + tid + 256) * DD;
            float a0 = sx[tid], a1 = sx[tid + 256];
            for (int dp = 0; dp < DD; ++dp) {
                float c = sctx[dp];
                a0 += c * ap0[dp];
                a1 += c * ap1[dp];
            }
            xp0 = a0; xp1 = a1;
        }
    } else {
        xp0 = sx[tid]; xp1 = sx[tid + 256];
    }
    __syncthreads();

    // 8. LayerNorm over D
    red[tid] = xp0 + xp1;
    __syncthreads();
    for (int off = 128; off > 0; off >>= 1) { if (tid < off) red[tid] += red[tid + off]; __syncthreads(); }
    float mu = red[0] * (1.0f / DD);
    __syncthreads();
    float v0 = xp0 - mu, v1 = xp1 - mu;
    red[tid] = v0 * v0 + v1 * v1;
    __syncthreads();
    for (int off = 128; off > 0; off >>= 1) { if (tid < off) red[tid] += red[tid + off]; __syncthreads(); }
    float rstd = rsqrtf(red[0] * (1.0f / DD) + 1e-5f);
    x[b * DD + tid]       = v0 * rstd * ln_g[l * DD + tid]       + ln_b[l * DD + tid];
    x[b * DD + tid + 256] = v1 * rstd * ln_g[l * DD + tid + 256] + ln_b[l * DD + tid + 256];
}

// ---------------------------------------------------------------------------
// Resonator part 1: cs[b,n] = sum_d cos(th), ss[b,n] = sum_d sin(th)
// (the dominant transcendental workload: B*N*D sin+cos per layer-step)
// ---------------------------------------------------------------------------
__global__ void res_sum_kernel(const float* __restrict__ res_W, const float* __restrict__ res_B,
                               const float* __restrict__ x,
                               float* __restrict__ cs, float* __restrict__ ss,
                               int l, int t, const int* __restrict__ tstart) {
    __shared__ float sxb[DD];
    const int tid = threadIdx.x;
    const int b = blockIdx.x >> 2;
    const int n = (blockIdx.x & 3) * 256 + tid;
    const float t_phi = (float)(t + *tstart) * PHI_F;
    for (int d = tid; d < DD; d += 256) sxb[d] = x[b * DD + d];
    __syncthreads();
    const float* rw = res_W + ((size_t)l * NN + n) * DD;
    const float* rb = res_B + ((size_t)l * NN + n) * DD;
    float c = 0.0f, s = 0.0f;
    for (int d = 0; d < DD; ++d) {
        float th = sxb[d] / (1.0f + fabsf(rw[d])) + rb[d] + t_phi;
        c += __cosf(th);
        s += __sinf(th);
    }
    cs[b * NN + n] = c;
    ss[b * NN + n] = s;
}

// ---------------------------------------------------------------------------
// Resonator part 2: r = cs @ res_real^T + ss @ res_imag^T ; x += silu(r)
// ---------------------------------------------------------------------------
__global__ void res_proj_kernel(const float* __restrict__ res_real, const float* __restrict__ res_imag,
                                const float* __restrict__ cs, const float* __restrict__ ss,
                                float* __restrict__ x, int l) {
    int idx = blockIdx.x * 256 + threadIdx.x;       // [0, B*D)
    if (idx >= BB * DD) return;
    int b = idx >> 9, d = idx & (DD - 1);
    const float* rr  = res_real + ((size_t)l * DD + d) * NN;
    const float* ri  = res_imag + ((size_t)l * DD + d) * NN;
    const float* csb = cs + b * NN;
    const float* ssb = ss + b * NN;
    float acc = 0.0f;
    for (int n = 0; n < NN; ++n) acc += csb[n] * rr[n] + ssb[n] * ri[n];
    float sig = 1.0f / (1.0f + __expf(-acc));
    x[idx] += acc * sig;
}

// ---------------------------------------------------------------------------
// Final LayerNorm + pack x into a zero-padded 16x512 fp16 A-tile for WMMA
// ---------------------------------------------------------------------------
__global__ void final_ln_kernel(const float* __restrict__ out_g, const float* __restrict__ out_b,
                                const float* __restrict__ x, _Float16* __restrict__ xpad) {
    __shared__ float red[256];
    const int tid = threadIdx.x;
    for (int b = 0; b < BB; ++b) {
        float v0 = x[b * DD + tid], v1 = x[b * DD + tid + 256];
        red[tid] = v0 + v1;
        __syncthreads();
        for (int off = 128; off > 0; off >>= 1) { if (tid < off) red[tid] += red[tid + off]; __syncthreads(); }
        float mu = red[0] * (1.0f / DD);
        __syncthreads();
        float d0 = v0 - mu, d1 = v1 - mu;
        red[tid] = d0 * d0 + d1 * d1;
        __syncthreads();
        for (int off = 128; off > 0; off >>= 1) { if (tid < off) red[tid] += red[tid + off]; __syncthreads(); }
        float rstd = rsqrtf(red[0] * (1.0f / DD) + 1e-5f);
        __syncthreads();
        xpad[b * DD + tid]       = (_Float16)(d0 * rstd * out_g[tid]       + out_b[tid]);
        xpad[b * DD + tid + 256] = (_Float16)(d1 * rstd * out_g[tid + 256] + out_b[tid + 256]);
    }
    // zero padding rows 2..15 so the WMMA M-dimension is well defined
    for (int i = tid; i < (16 - BB) * DD; i += 256) xpad[BB * DD + i] = (_Float16)0.0f;
}

// ---------------------------------------------------------------------------
// Logits GEMM: (16 x 512) fp16 A-tile  x  out_W^T tiles  via
// v_wmma_f32_16x16x32_f16. One wave per 16-vocab tile, 8 waves/block,
// 250 blocks covering V=32000. A staged in LDS, W streamed from L2.
// ---------------------------------------------------------------------------
__global__ void logits_wmma_kernel(const _Float16* __restrict__ Wh,
                                   const _Float16* __restrict__ xpad,
                                   float* __restrict__ out, int t) {
    __shared__ __align__(16) _Float16 shx[16 * DD];   // 16KB A tile
    // cooperative load of the A tile into LDS
    {
        const uint32_t* src = (const uint32_t*)xpad;
        uint32_t* dst = (uint32_t*)shx;
        for (int i = threadIdx.x; i < 16 * DD / 2; i += 256) dst[i] = src[i];
    }
    __syncthreads();

    const int wave = threadIdx.x >> 5;
    const int lane = threadIdx.x & 31;
    const int tile = blockIdx.x * 8 + wave;
    const int v0 = tile * 16;

    const int row   = lane & 15;                 // A row (M) / B column (N)
    const int koffA = (lane < 16) ? 0 : 8;       // 16-bit A layout: {0-7,16-23} / {8-15,24-31}
    const int koffB = (lane < 16) ? 0 : 8;       // 16-bit B layout: {0-7,16-23} / {8-15,24-31}

    const _Float16* wrow = Wh + (size_t)(v0 + row) * DD;

    v8f c = {};
    #pragma unroll
    for (int k0 = 0; k0 < DD; k0 += 32) {
        v8h a0 = *(const v8h*)(&shx[row * DD + k0 + koffA]);
        v8h a1 = *(const v8h*)(&shx[row * DD + k0 + koffA + 16]);
        v16h a;
        #pragma unroll
        for (int i = 0; i < 8; ++i) { a[i] = a0[i]; a[i + 8] = a1[i]; }
        v8h b0 = *(const v8h*)(&wrow[k0 + koffB]);
        v8h b1 = *(const v8h*)(&wrow[k0 + koffB + 16]);
        v16h bfrag;
        #pragma unroll
        for (int i = 0; i < 8; ++i) { bfrag[i] = b0[i]; bfrag[i + 8] = b1[i]; }
        c = __builtin_amdgcn_wmma_f32_16x16x32_f16(false, a, false, bfrag,
                                                   (short)0, c, false, false);
    }

    // C/D layout: VGPR i holds M=i for lanes 0-15 (N=lane). Only M=0,1 valid.
    if (lane < 16) {
        int v = v0 + lane;
        out[(size_t)t * VV + v]                     = c[0];   // batch 0
        out[(size_t)SS * VV + (size_t)t * VV + v]   = c[1];   // batch 1
    }
}

// ---------------------------------------------------------------------------
// Host driver
// ---------------------------------------------------------------------------
extern "C" void kernel_launch(void* const* d_in, const int* in_sizes, int n_in,
                              void* d_out, int out_size, void* d_ws, size_t ws_size,
                              hipStream_t stream) {
    (void)in_sizes; (void)n_in; (void)out_size; (void)ws_size;

    const int*   ids       = (const int*)  d_in[0];
    const float* temb      = (const float*)d_in[1];
    const float* wq        = (const float*)d_in[2];
    const float* bq        = (const float*)d_in[3];
    const float* wk        = (const float*)d_in[4];
    const float* bk        = (const float*)d_in[5];
    const float* attn_proj = (const float*)d_in[6];
    const float* res_W     = (const float*)d_in[7];
    const float* res_B     = (const float*)d_in[8];
    const float* res_real  = (const float*)d_in[9];
    const float* res_imag  = (const float*)d_in[10];
    const float* ln_g      = (const float*)d_in[11];
    const float* ln_b      = (const float*)d_in[12];
    const float* out_g     = (const float*)d_in[13];
    const float* out_b     = (const float*)d_in[14];
    const float* out_W     = (const float*)d_in[15];
    const int*   tstart    = (const int*)  d_in[16];
    float* out = (float*)d_out;

    // workspace layout
    char* ws = (char*)d_ws;
    _Float16* Wh    = (_Float16*)ws;                                   // 32,768,000 B
    float*    cache = (float*)(ws + (size_t)VV * DD * 2);              //  2,097,152 B
    float*    hre   = cache + (size_t)LL * BB * SS * DD;               //  1024 f
    float*    him   = hre + BB * DD;                                   //  1024 f
    float*    xbuf  = him + BB * DD;                                   //  1024 f
    float*    cs    = xbuf + BB * DD;                                  //  2048 f
    float*    ssum  = cs + BB * NN;                                    //  2048 f
    _Float16* xpad  = (_Float16*)(ssum + BB * NN);                     //  16*512 f16

    // one-time prep
    convert_w_kernel<<<(VV * DD + 255) / 256, 256, 0, stream>>>(out_W, Wh, VV * DD);
    init_zero_kernel<<<(2 * BB * DD + 255) / 256, 256, 0, stream>>>(hre, 2 * BB * DD); // hre+him

    for (int t = 0; t < SS; ++t) {
        osc_kernel<<<(BB * DD + 255) / 256, 256, 0, stream>>>(ids, temb, hre, him, xbuf, t, tstart);
        for (int l = 0; l < LL; ++l) {
            attn_kernel<<<BB, 256, 0, stream>>>(wq, bq, wk, bk, attn_proj, ln_g, ln_b,
                                                xbuf, cache, l, t, tstart);
            res_sum_kernel<<<BB * (NN / 256), 256, 0, stream>>>(res_W, res_B, xbuf, cs, ssum,
                                                                l, t, tstart);
            res_proj_kernel<<<(BB * DD + 255) / 256, 256, 0, stream>>>(res_real, res_imag,
                                                                       cs, ssum, xbuf, l);
        }
        final_ln_kernel<<<1, 256, 0, stream>>>(out_g, out_b, xbuf, xpad);
        logits_wmma_kernel<<<VV / (16 * 8), 256, 0, stream>>>(Wh, xpad, out, t);
    }
}
